// RNNSequenceEncoder_46016279609932
// MI455X (gfx1250) — matrix operand order repaired and checked by reading, not compiled, not run on hardware
//
#include <hip/hip_runtime.h>

// RNN sequence encoder for MI455X (gfx1250, wave32, WMMA bf16 16x16x32).
// Persistent per-batch-chunk workgroups: batch rows are independent through
// the scan, so each of 16 WGs (1024 threads = 32 waves) evolves 16 batch rows
// across all 256 timesteps with only workgroup barriers. Weights pre-packed
// into WMMA B-operand layout (bf16); state kept in f32 in LDS; f32 accumulate.

#define B_   256
#define T_   256
#define IN_  128
#define H_   1024
#define BC   16     // batch rows per workgroup
#define NT   2      // 16-wide n-tiles per wave (32 waves * 32 cols = 1024 = H)

typedef __attribute__((ext_vector_type(16))) __bf16 v16bf;
typedef __attribute__((ext_vector_type(8)))  __bf16 v8bf;
typedef __attribute__((ext_vector_type(8)))  float  v8f;
typedef __attribute__((ext_vector_type(4)))  float  v4f;

// Pack W[N][K] (f32, K contiguous) into WMMA-B tiles, bf16.
// Tile (nb,kb): 32 lanes x 16 bf16. lane l: n = nb*16+(l&15),
// k0 = kb*32 + (l>>4)*8, element j holds W[n][k0+j] for j<8 and
// W[n][k0+16+(j-8)] for j>=8 (ISA table order).
__global__ void pack_w(const float* __restrict__ W, __bf16* __restrict__ P,
                       int N, int K) {
  int gid = blockIdx.x * blockDim.x + threadIdx.x;
  int total = (N * K) >> 4;
  if (gid >= total) return;
  int lane = gid & 31;
  int tile = gid >> 5;
  int KB = K >> 5;
  int kb = tile % KB;
  int nb = tile / KB;
  int n  = (nb << 4) + (lane & 15);
  int k0 = (kb << 5) + ((lane >> 4) << 3);
  const float* src = W + (size_t)n * K + k0;
  __bf16* dst = P + ((size_t)gid << 4);
#pragma unroll
  for (int j = 0; j < 16; ++j) dst[j] = (__bf16)src[j + ((j >> 3) << 3)];
}

__launch_bounds__(1024, 1)
__global__ void rnn_scan(const float* __restrict__ x,
                         const float* __restrict__ b_in,
                         const float* __restrict__ b_rec,
                         const float* __restrict__ b_out,
                         const __bf16* __restrict__ Pin,
                         const __bf16* __restrict__ Prec,
                         const __bf16* __restrict__ Pout,
                         float* __restrict__ out) {
  __shared__ float  s_state[BC][H_ + 8];    // f32 state, padded rows
  __shared__ __bf16 s_abuf[BC][H_ + 16];    // bf16(state + ext) = WMMA A source
  __shared__ float  s_x[BC][IN_ + 8];       // staged x[:, t, :] tile

  const int tid   = threadIdx.x;
  const int lane  = tid & 31;
  const int wave  = tid >> 5;
  const int l16   = lane & 15;
  const int kh    = lane >> 4;              // K-half select per ISA A/B layout
  const int bbase = blockIdx.x * BC;
  const int nb0   = wave * NT;

  for (int idx = tid; idx < BC * H_; idx += 1024)
    s_state[idx >> 10][idx & (H_ - 1)] = 0.f;

  float bi[NT], br[NT];
#pragma unroll
  for (int j = 0; j < NT; ++j) {
    int n = ((nb0 + j) << 4) + l16;
    bi[j] = b_in[n];
    br[j] = b_rec[n];
  }
  __syncthreads();

  const v16bf* PinV  = (const v16bf*)Pin;
  const v16bf* PrecV = (const v16bf*)Prec;
  const v16bf* PoutV = (const v16bf*)Pout;
  const v8f z8 = {0.f, 0.f, 0.f, 0.f, 0.f, 0.f, 0.f, 0.f};

  for (int t = 0; t < T_; ++t) {
    // stage x[bbase..bbase+16, t, 0..128) -> LDS (coalesced)
    for (int idx = tid; idx < BC * IN_; idx += 1024) {
      int m = idx >> 7, i = idx & (IN_ - 1);
      s_x[m][i] = x[((size_t)(bbase + m) * T_ + t) * IN_ + i];
    }
    __syncthreads();

    // ---- phase E: ext = x_t @ W_in^T (K=128, bf16 WMMA) ----
    v8f e[NT];
#pragma unroll
    for (int j = 0; j < NT; ++j) e[j] = z8;
#pragma unroll
    for (int kb = 0; kb < (IN_ >> 5); ++kb) {
      int c1 = (kb << 5) + (kh << 3);
      v8f f1 = *(const v8f*)&s_x[l16][c1];
      v8f f2 = *(const v8f*)&s_x[l16][c1 + 16];
      v16bf a;
#pragma unroll
      for (int i2 = 0; i2 < 8; ++i2) { a[i2] = (__bf16)f1[i2]; a[i2 + 8] = (__bf16)f2[i2]; }
#pragma unroll
      for (int j = 0; j < NT; ++j) {
        v16bf bb = PinV[(size_t)((nb0 + j) * (IN_ >> 5) + kb) * 32 + lane];
        e[j] = __builtin_amdgcn_wmma_f32_16x16x32_bf16(false, a, false, bb,
                                                       (short)0, e[j], false, false);
      }
    }
    // abuf = bf16(state + ext + b_in)   (C/D layout: elem r -> m=r+8*kh, n=l16)
#pragma unroll
    for (int j = 0; j < NT; ++j) {
      int n = ((nb0 + j) << 4) + l16;
#pragma unroll
      for (int r = 0; r < 8; ++r) {
        int m = r + (kh << 3);
        s_abuf[m][n] = (__bf16)(s_state[m][n] + e[j][r] + bi[j]);
      }
    }
    __syncthreads();

    // ---- phase R: new_state = relu(abuf @ W_rec^T + b_rec) (K=1024) ----
    v8f acc[NT];
#pragma unroll
    for (int j = 0; j < NT; ++j) acc[j] = z8;
#pragma unroll 4
    for (int kb = 0; kb < (H_ >> 5); ++kb) {
      int c1 = (kb << 5) + (kh << 3);
      v8bf lo = *(const v8bf*)&s_abuf[l16][c1];
      v8bf hi = *(const v8bf*)&s_abuf[l16][c1 + 16];
      v16bf a;
#pragma unroll
      for (int i2 = 0; i2 < 8; ++i2) { a[i2] = lo[i2]; a[i2 + 8] = hi[i2]; }
#pragma unroll
      for (int j = 0; j < NT; ++j) {
        v16bf bb = PrecV[(size_t)((nb0 + j) * (H_ >> 5) + kb) * 32 + lane];
        acc[j] = __builtin_amdgcn_wmma_f32_16x16x32_bf16(false, a, false, bb,
                                                         (short)0, acc[j], false, false);
      }
    }
#pragma unroll
    for (int j = 0; j < NT; ++j) {
      int n = ((nb0 + j) << 4) + l16;
#pragma unroll
      for (int r = 0; r < 8; ++r) {
        int m = r + (kh << 3);
        float v = acc[j][r] + br[j];
        s_state[m][n] = v > 0.f ? v : 0.f;
      }
    }
    __syncthreads();
  }

  // ---- final: plop = state @ W_out^T + b_out, broadcast over T ----
#pragma unroll
  for (int j = 0; j < NT; ++j) {
    int n = ((nb0 + j) << 4) + l16;
#pragma unroll
    for (int r = 0; r < 8; ++r) {
      int m = r + (kh << 3);
      s_abuf[m][n] = (__bf16)s_state[m][n];
    }
  }
  __syncthreads();

  v8f p[NT];
#pragma unroll
  for (int j = 0; j < NT; ++j) p[j] = z8;
#pragma unroll 4
  for (int kb = 0; kb < (H_ >> 5); ++kb) {
    int c1 = (kb << 5) + (kh << 3);
    v8bf lo = *(const v8bf*)&s_abuf[l16][c1];
    v8bf hi = *(const v8bf*)&s_abuf[l16][c1 + 16];
    v16bf a;
#pragma unroll
    for (int i2 = 0; i2 < 8; ++i2) { a[i2] = lo[i2]; a[i2 + 8] = hi[i2]; }
#pragma unroll
    for (int j = 0; j < NT; ++j) {
      v16bf bb = PoutV[(size_t)((nb0 + j) * (H_ >> 5) + kb) * 32 + lane];
      p[j] = __builtin_amdgcn_wmma_f32_16x16x32_bf16(false, a, false, bb,
                                                     (short)0, p[j], false, false);
    }
  }
#pragma unroll
  for (int j = 0; j < NT; ++j) {
    int n = ((nb0 + j) << 4) + l16;
    float bo = b_out[n];
#pragma unroll
    for (int r = 0; r < 8; ++r) {
      int m = r + (kh << 3);
      s_state[m][n] = p[j][r] + bo;
    }
  }
  __syncthreads();

  // out[b, t, :] = plop[b, :] for all t  (coalesced 16B stores)
  for (int m = 0; m < BC; ++m) {
    const v4f* row = (const v4f*)&s_state[m][0];
    for (int i = tid; i < T_ * (H_ >> 2); i += 1024) {
      int tt = i >> 8;
      int c  = i & ((H_ >> 2) - 1);
      ((v4f*)out)[((size_t)(bbase + m) * T_ + tt) * (H_ >> 2) + c] = row[c];
    }
  }
}

extern "C" void kernel_launch(void* const* d_in, const int* in_sizes, int n_in,
                              void* d_out, int out_size, void* d_ws, size_t ws_size,
                              hipStream_t stream) {
  const float* x     = (const float*)d_in[0];
  const float* W_in  = (const float*)d_in[1];
  const float* b_in  = (const float*)d_in[2];
  const float* W_rec = (const float*)d_in[3];
  const float* b_rec = (const float*)d_in[4];
  const float* W_out = (const float*)d_in[5];
  const float* b_out = (const float*)d_in[6];
  float* out = (float*)d_out;

  // ws: packed bf16 weights (256KB + 2MB + 2MB = 4.25MB)
  __bf16* Pin  = (__bf16*)d_ws;
  __bf16* Prec = Pin  + (size_t)H_ * IN_;
  __bf16* Pout = Prec + (size_t)H_ * H_;

  pack_w<<<(H_ * IN_ / 16 + 255) / 256, 256, 0, stream>>>(W_in,  Pin,  H_, IN_);
  pack_w<<<(H_ * H_  / 16 + 255) / 256, 256, 0, stream>>>(W_rec, Prec, H_, H_);
  pack_w<<<(H_ * H_  / 16 + 255) / 256, 256, 0, stream>>>(W_out, Pout, H_, H_);

  rnn_scan<<<B_ / BC, 1024, 0, stream>>>(x, b_in, b_rec, b_out,
                                         Pin, Prec, Pout, out);
}
